// BIMPM_37082747633676
// MI455X (gfx1250) — compile-verified, matrix operand order checked by reading, not run on hardware
//
#include <hip/hip_runtime.h>
#include <math.h>

#define EPSF 1e-6f

typedef __attribute__((ext_vector_type(16))) __bf16 v16bf;
typedef __attribute__((ext_vector_type(8)))  float  v8f;

__device__ __forceinline__ float sigmoidf(float x) { return 1.0f / (1.0f + expf(-x)); }

// ---------------------------------------------------------------------------
// Generic batched WMMA GEMM:  C[m,n] = act( sum_k A[m,k]*B(n,k) + bias[n] + Cin[m,n] )
// A: row-major [M,K] (lda). B: if BT==0, [N,K] row-major (weights [out,in]);
// if BT==1, element (n,k) = Bsrc[k*ldb + n] (i.e. [K,N] row-major source).
// One wave (32 threads) per 16x64 tile (4 WMMA accumulators). bf16 in, f32 acc.
// Requires: M%16==0, N%64==0, even lda/ldb. K arbitrary (guarded tail chunk).
// ---------------------------------------------------------------------------
template <int BT>
__global__ void gemm_bf16_wmma(
    const float* __restrict__ A, int lda, long long batchA,
    const float* __restrict__ Bm, int ldb, long long batchB,
    const float* __restrict__ bias,
    const float* __restrict__ Cin, int ldcin, long long batchCin,
    float* __restrict__ C, int ldc, long long batchC,
    int M, int N, int K, int act)
{
    const int lane = threadIdx.x & 31;
    const int n0 = blockIdx.x << 6;     // 64 columns per wave
    const int m0 = blockIdx.y << 4;
    const int bz = blockIdx.z;
    A  += (long long)bz * batchA;
    Bm += (long long)bz * batchB;
    if (Cin) Cin += (long long)bz * batchCin;
    C  += (long long)bz * batchC;

    const int row = lane & 15;          // M row (A) / N col (B,C/D)
    const int kh  = lane >> 4;          // K-half for A/B frags, M-half for C/D

    v8f acc[4];
    if (Cin) {
        #pragma unroll
        for (int jt = 0; jt < 4; ++jt) {
            int nn = n0 + jt * 16 + row;
            #pragma unroll
            for (int r = 0; r < 8; ++r)
                acc[jt][r] = Cin[(long long)(m0 + r + (kh << 3)) * ldcin + nn];
        }
    } else {
        #pragma unroll
        for (int jt = 0; jt < 4; ++jt)
            #pragma unroll
            for (int r = 0; r < 8; ++r) acc[jt][r] = 0.0f;
    }

    const float* Arow = A + (long long)(m0 + row) * lda;
    int kb = 0;
    // ---- main K loop: unguarded vector loads ----
    for (; kb + 32 <= K; kb += 32) {
        v16bf af;
        #pragma unroll
        for (int v = 0; v < 8; ++v) {
            int k0 = kb + ((v >> 2) << 4) + (kh << 3) + ((v & 3) << 1);
            float2 a2 = *(const float2*)(Arow + k0);
            af[2 * v]     = (__bf16)a2.x;
            af[2 * v + 1] = (__bf16)a2.y;
        }
        #pragma unroll
        for (int jt = 0; jt < 4; ++jt) {
            int nn = n0 + jt * 16 + row;
            v16bf bfm;
            #pragma unroll
            for (int v = 0; v < 8; ++v) {
                int k0 = kb + ((v >> 2) << 4) + (kh << 3) + ((v & 3) << 1);
                if (BT) {
                    bfm[2 * v]     = (__bf16)Bm[(long long)k0 * ldb + nn];
                    bfm[2 * v + 1] = (__bf16)Bm[(long long)(k0 + 1) * ldb + nn];
                } else {
                    float2 b2 = *(const float2*)(Bm + (long long)nn * ldb + k0);
                    bfm[2 * v]     = (__bf16)b2.x;
                    bfm[2 * v + 1] = (__bf16)b2.y;
                }
            }
            acc[jt] = __builtin_amdgcn_wmma_f32_16x16x32_bf16(false, af, false, bfm,
                                                              (short)0, acc[jt], false, false);
        }
    }
    // ---- guarded tail chunk (only K%32 != 0, e.g. K=364) ----
    if (kb < K) {
        v16bf af;
        #pragma unroll
        for (int v = 0; v < 8; ++v) {
            int k0 = kb + ((v >> 2) << 4) + (kh << 3) + ((v & 3) << 1);
            af[2 * v]     = (__bf16)((k0     < K) ? Arow[k0]     : 0.0f);
            af[2 * v + 1] = (__bf16)((k0 + 1 < K) ? Arow[k0 + 1] : 0.0f);
        }
        #pragma unroll
        for (int jt = 0; jt < 4; ++jt) {
            int nn = n0 + jt * 16 + row;
            v16bf bfm;
            #pragma unroll
            for (int v = 0; v < 8; ++v) {
                int k0 = kb + ((v >> 2) << 4) + (kh << 3) + ((v & 3) << 1);
                float b0, b1;
                if (BT) {
                    b0 = (k0     < K) ? Bm[(long long)k0 * ldb + nn]       : 0.0f;
                    b1 = (k0 + 1 < K) ? Bm[(long long)(k0 + 1) * ldb + nn] : 0.0f;
                } else {
                    const float* Brow = Bm + (long long)nn * ldb;
                    b0 = (k0     < K) ? Brow[k0]     : 0.0f;
                    b1 = (k0 + 1 < K) ? Brow[k0 + 1] : 0.0f;
                }
                bfm[2 * v]     = (__bf16)b0;
                bfm[2 * v + 1] = (__bf16)b1;
            }
            acc[jt] = __builtin_amdgcn_wmma_f32_16x16x32_bf16(false, af, false, bfm,
                                                              (short)0, acc[jt], false, false);
        }
    }

    #pragma unroll
    for (int jt = 0; jt < 4; ++jt) {
        int nn = n0 + jt * 16 + row;
        float bv = bias ? bias[nn] : 0.0f;
        #pragma unroll
        for (int r = 0; r < 8; ++r) {
            int mm = m0 + r + (kh << 3);
            float v = acc[jt][r] + bv;
            if (act == 1) v = tanhf(v);
            C[(long long)mm * ldc + nn] = v;
        }
    }
}

// ---------------------------------------------------------------------------
// LSTM cell pointwise: g=[i,f,g,o] chunks of Hh; updates h,c; optional output.
// ---------------------------------------------------------------------------
__global__ void lstm_cell(const float* __restrict__ g, float* __restrict__ h,
                          float* __restrict__ c, int N, int Hh,
                          float* __restrict__ out, long long outStride)
{
    int t = blockIdx.x * blockDim.x + threadIdx.x;
    if (t >= N * Hh) return;
    int n = t / Hh, j = t % Hh;
    const float* gr = g + (long long)n * 4 * Hh;
    float gi = gr[j], gf = gr[Hh + j], gg = gr[2 * Hh + j], go = gr[3 * Hh + j];
    float cv = sigmoidf(gf) * c[t] + sigmoidf(gi) * tanhf(gg);
    float hv = sigmoidf(go) * tanhf(cv);
    c[t] = cv; h[t] = hv;
    if (out) out[(long long)n * outStride + j] = hv;
}

__global__ void gather_rows(const float* __restrict__ emb, const int* __restrict__ idx,
                            float* __restrict__ dst, int nrows, int dim, int dstStride)
{
    long long t = (long long)blockIdx.x * blockDim.x + threadIdx.x;
    if (t >= (long long)nrows * dim) return;
    int r = (int)(t / dim), d = (int)(t % dim);
    dst[(long long)r * dstStride + d] = emb[(long long)idx[r] * dim + d];
}

__global__ void copy_strided(const float* __restrict__ src, int srcStride,
                             float* __restrict__ dst, int dstStride, int dstOff,
                             int nrows, int dim)
{
    int t = blockIdx.x * blockDim.x + threadIdx.x;
    if (t >= nrows * dim) return;
    int r = t / dim, d = t % dim;
    dst[(long long)r * dstStride + dstOff + d] = src[(long long)r * srcStride + d];
}

__global__ void row_norm(const float* __restrict__ x, float* __restrict__ nrm,
                         int nrows, int dim)
{
    int r = blockIdx.x * blockDim.x + threadIdx.x;
    if (r >= nrows) return;
    const float* xr = x + (long long)r * dim;
    float s = 0.0f;
    for (int d = 0; d < dim; ++d) s += xr[d] * xr[d];
    nrm[r] = sqrtf(fmaxf(s, EPSF));
}

__global__ void cm_normalize(float* __restrict__ cm, const float* __restrict__ n1,
                             const float* __restrict__ n2)
{
    int t = blockIdx.x * blockDim.x + threadIdx.x;
    if (t >= 64 * 64 * 64) return;
    int j = t & 63, bi = t >> 6, b = bi >> 6;
    cm[t] = cm[t] / (n1[bi] * n2[b * 64 + j]);
}

__global__ void cm_rowsum(const float* __restrict__ cm, float* __restrict__ rs,
                          int nrows, int ncols)
{
    int r = blockIdx.x * blockDim.x + threadIdx.x;
    if (r >= nrows) return;
    const float* c = cm + (long long)r * ncols;
    float s = 0.0f;
    for (int j = 0; j < ncols; ++j) s += c[j];
    rs[r] = s;
}

__global__ void cm_argmax(const float* __restrict__ cm, int* __restrict__ idx,
                          int nrows, int ncols)
{
    int r = blockIdx.x * blockDim.x + threadIdx.x;
    if (r >= nrows) return;
    const float* c = cm + (long long)r * ncols;
    float best = c[0]; int bi = 0;
    for (int j = 1; j < ncols; ++j) { if (c[j] > best) { best = c[j]; bi = j; } }
    idx[r] = bi;
}

// full-matching: out[b,l,off+p] = cos(h1[b,l]*w[p], h2[b,last]*w[p])
__global__ void mpm_full(const float* __restrict__ h1, const float* __restrict__ h2,
                         const float* __restrict__ w, float* __restrict__ out, int outOff)
{
    int t = blockIdx.x * blockDim.x + threadIdx.x;
    if (t >= 64 * 64 * 20) return;
    int p = t % 20, bl = t / 20, b = bl / 64;
    const float* x1 = h1 + (long long)bl * 128;
    const float* x2 = h2 + ((long long)b * 64 + 63) * 128;
    const float* wp = w + p * 128;
    float num = 0, s1 = 0, s2 = 0;
    for (int d = 0; d < 128; ++d) {
        float a = x1[d] * wp[d], c = x2[d] * wp[d];
        num += a * c; s1 += a * a; s2 += c * c;
    }
    out[(long long)bl * 160 + outOff + p] =
        num / (sqrtf(fmaxf(s1, EPSF)) * sqrtf(fmaxf(s2, EPSF)));
}

// attentive: att row = attraw[b,l]/(rowsum+eps); cosine vs h1*w
__global__ void mpm_att(const float* __restrict__ h1, const float* __restrict__ att,
                        const float* __restrict__ rs, const float* __restrict__ w,
                        float* __restrict__ out, int outOff)
{
    int t = blockIdx.x * blockDim.x + threadIdx.x;
    if (t >= 64 * 64 * 20) return;
    int p = t % 20, bl = t / 20;
    const float* x1 = h1 + (long long)bl * 128;
    const float* x2 = att + (long long)bl * 128;
    const float* wp = w + p * 128;
    float inv = 1.0f / (rs[bl] + EPSF);
    float num = 0, s1 = 0, s2 = 0;
    for (int d = 0; d < 128; ++d) {
        float a = x1[d] * wp[d], c = x2[d] * inv * wp[d];
        num += a * c; s1 += a * a; s2 += c * c;
    }
    out[(long long)bl * 160 + outOff + p] =
        num / (sqrtf(fmaxf(s1, EPSF)) * sqrtf(fmaxf(s2, EPSF)));
}

__global__ void mpm_maxatt(const float* __restrict__ h1, const float* __restrict__ h2,
                           const int* __restrict__ idx, const float* __restrict__ w,
                           float* __restrict__ out, int outOff)
{
    int t = blockIdx.x * blockDim.x + threadIdx.x;
    if (t >= 64 * 64 * 20) return;
    int p = t % 20, bl = t / 20, b = bl / 64;
    const float* x1 = h1 + (long long)bl * 128;
    const float* x2 = h2 + ((long long)b * 64 + idx[bl]) * 128;
    const float* wp = w + p * 128;
    float num = 0, s1 = 0, s2 = 0;
    for (int d = 0; d < 128; ++d) {
        float a = x1[d] * wp[d], c = x2[d] * wp[d];
        num += a * c; s1 += a * a; s2 += c * c;
    }
    out[(long long)bl * 160 + outOff + p] =
        num / (sqrtf(fmaxf(s1, EPSF)) * sqrtf(fmaxf(s2, EPSF)));
}

// norms of h[b,i]*w[p] -> o[(b*64+i)*20+p]
__global__ void norms_hw(const float* __restrict__ h, const float* __restrict__ w,
                         float* __restrict__ o)
{
    int t = blockIdx.x * blockDim.x + threadIdx.x;
    if (t >= 64 * 64 * 20) return;
    int p = t % 20, bl = t / 20;
    const float* x = h + (long long)bl * 128;
    const float* wp = w + p * 128;
    float s = 0.0f;
    for (int d = 0; d < 128; ++d) { float a = x[d] * wp[d]; s += a * a; }
    o[t] = sqrtf(fmaxf(s, EPSF));
}

// max-pool matching: per (b,p), wave computes 16 rows x all 64 cols of
// cos((h1*w),(h2*w)) via WMMA over K=128 and reduces max over cols.
__global__ void maxpool_wmma_k(const float* __restrict__ h1, const float* __restrict__ h2,
                               const float* __restrict__ w,
                               const float* __restrict__ n1w, const float* __restrict__ n2w,
                               float* __restrict__ mout, int outOff)
{
    const int lane = threadIdx.x & 31;
    const int it = blockIdx.x, p = blockIdx.y, b = blockIdx.z;
    const float* h1b = h1 + (long long)b * 64 * 128;
    const float* h2b = h2 + (long long)b * 64 * 128;
    const float* wp  = w + p * 128;
    const int row = lane & 15;
    const int kh  = lane >> 4;

    v8f acc[4];
    #pragma unroll
    for (int jt = 0; jt < 4; ++jt)
        #pragma unroll
        for (int r = 0; r < 8; ++r) acc[jt][r] = 0.0f;

    for (int kb = 0; kb < 128; kb += 32) {
        v16bf af;
        #pragma unroll
        for (int v = 0; v < 8; ++v) {
            int k0 = kb + ((v >> 2) << 4) + (kh << 3) + ((v & 3) << 1);
            float2 a2 = *(const float2*)(h1b + (it * 16 + row) * 128 + k0);
            float2 w2 = *(const float2*)(wp + k0);
            af[2 * v]     = (__bf16)(a2.x * w2.x);
            af[2 * v + 1] = (__bf16)(a2.y * w2.y);
        }
        #pragma unroll
        for (int jt = 0; jt < 4; ++jt) {
            v16bf bfm;
            #pragma unroll
            for (int v = 0; v < 8; ++v) {
                int k0 = kb + ((v >> 2) << 4) + (kh << 3) + ((v & 3) << 1);
                float2 b2 = *(const float2*)(h2b + (jt * 16 + row) * 128 + k0);
                float2 w2 = *(const float2*)(wp + k0);
                bfm[2 * v]     = (__bf16)(b2.x * w2.x);
                bfm[2 * v + 1] = (__bf16)(b2.y * w2.y);
            }
            acc[jt] = __builtin_amdgcn_wmma_f32_16x16x32_bf16(false, af, false, bfm,
                                                              (short)0, acc[jt], false, false);
        }
    }

    float mx[8];
    #pragma unroll
    for (int r = 0; r < 8; ++r) {
        float m = -3.4e38f;
        #pragma unroll
        for (int jt = 0; jt < 4; ++jt) {
            int j = jt * 16 + row;
            m = fmaxf(m, acc[jt][r] / n2w[((long long)b * 64 + j) * 20 + p]);
        }
        mx[r] = m;
    }
    #pragma unroll
    for (int r = 0; r < 8; ++r) {
        #pragma unroll
        for (int off = 1; off < 16; off <<= 1)
            mx[r] = fmaxf(mx[r], __shfl_xor(mx[r], off, 32));
    }
    if (row == 0) {
        #pragma unroll
        for (int r = 0; r < 8; ++r) {
            int m = it * 16 + r + (kh << 3);
            mout[((long long)b * 64 + m) * 160 + outOff + p] =
                mx[r] / n1w[((long long)b * 64 + m) * 20 + p];
        }
    }
}

__global__ void fc2_k(const float* __restrict__ x, const float* __restrict__ W,
                      const float* __restrict__ bias, float* __restrict__ out)
{
    int t = blockIdx.x * blockDim.x + threadIdx.x;
    if (t >= 128) return;
    int b = t >> 1, o = t & 1;
    float s = bias[o];
    const float* xr = x + b * 256;
    const float* wr = W + o * 256;
    for (int k = 0; k < 256; ++k) s += xr[k] * wr[k];
    out[t] = s;
}

// ---------------------------------------------------------------------------
static inline void launch_gemm(hipStream_t s,
    const float* A, int lda, long long bA,
    const float* Bm, int ldb, long long bB, int btrans,
    const float* bias, const float* Cin, int ldcin, long long bCin,
    float* C, int ldc, long long bC, int M, int N, int K, int act, int batch)
{
    dim3 grid(N / 64, M / 16, batch);
    if (btrans)
        gemm_bf16_wmma<1><<<grid, 32, 0, s>>>(A, lda, bA, Bm, ldb, bB,
                                              bias, Cin, ldcin, bCin, C, ldc, bC, M, N, K, act);
    else
        gemm_bf16_wmma<0><<<grid, 32, 0, s>>>(A, lda, bA, Bm, ldb, bB,
                                              bias, Cin, ldcin, bCin, C, ldc, bC, M, N, K, act);
}

extern "C" void kernel_launch(void* const* d_in, const int* in_sizes, int n_in,
                              void* d_out, int out_size, void* d_ws, size_t ws_size,
                              hipStream_t stream)
{
    const int*   idxA  = (const int*)d_in[0];
    const int*   idxBc = (const int*)d_in[1];
    const int*   cidxA = (const int*)d_in[2];
    const int*   cidxB = (const int*)d_in[3];
    const float* word_emb = (const float*)d_in[4];
    const float* char_emb = (const float*)d_in[5];
    const float* cWih = (const float*)d_in[6];
    const float* cWhh = (const float*)d_in[7];
    const float* cb   = (const float*)d_in[8];
    const float* xWf  = (const float*)d_in[9];
    const float* xUf  = (const float*)d_in[10];
    const float* xbf  = (const float*)d_in[11];
    const float* xWb  = (const float*)d_in[12];
    const float* xUb  = (const float*)d_in[13];
    const float* xbb  = (const float*)d_in[14];
    const float* aWf  = (const float*)d_in[15];
    const float* aUf  = (const float*)d_in[16];
    const float* abf  = (const float*)d_in[17];
    const float* aWb  = (const float*)d_in[18];
    const float* aUb  = (const float*)d_in[19];
    const float* abb  = (const float*)d_in[20];
    const float* mpmw = (const float*)d_in[21];
    const float* fc1W = (const float*)d_in[22];
    const float* fc1b = (const float*)d_in[23];
    const float* fc2W = (const float*)d_in[24];
    const float* fc2b = (const float*)d_in[25];
    float* out = (float*)d_out;

    // --- workspace layout (floats) ---
    float* W0 = (float*)d_ws;
    size_t off = 0;
    auto alloc = [&](size_t n) { float* p = W0 + off; off += n; return p; };
    float* bufX = alloc(4096 * 364);        // [B*L, 364] word||char
    float* cemb = alloc(4096 * 8 * 64);     // char embeddings
    float* R1   = alloc(8388608);           // cxw (32768x256) / xwF+xwB (2x 4096x512)
    float* gbuf = alloc(4096 * 256);        // gate buffer
    float* hst  = alloc(4096 * 64);         // hidden state (max char size)
    float* cst  = alloc(4096 * 64);         // cell state
    float* p_fw = alloc(4096 * 128);
    float* p_bw = alloc(4096 * 128);
    float* q_fw = alloc(4096 * 128);
    float* q_bw = alloc(4096 * 128);
    float* cmF  = alloc(64 * 64 * 64);
    float* cmB  = alloc(64 * 64 * 64);
    float* nPF = alloc(4096); float* nPB = alloc(4096);
    float* nQF = alloc(4096); float* nQB = alloc(4096);
    float* rsF = alloc(4096); float* rsB = alloc(4096);
    int* aidxF = (int*)alloc(4096);
    int* aidxB = (int*)alloc(4096);
    float* attb = alloc(4096 * 128);
    float* n1w = alloc(64 * 64 * 20);
    float* n2w = alloc(64 * 64 * 20);
    float* mp  = alloc(4096 * 160);
    float* mq  = alloc(4096 * 160);
    float* fcx  = alloc(64 * 512);
    float* fc1o = alloc(64 * 256);
    float* cxw = R1;
    float* xwF = R1;
    float* xwB = R1 + 2097152;

    auto blocks = [](long long n) { return (unsigned)((n + 255) / 256); };

    // ------------------- encoder: word/char emb + char LSTM + ctx BiLSTM ----
    auto encode = [&](const int* widx, const int* cidx, float* o_fw, float* o_bw) {
        gather_rows<<<blocks(4096LL * 300), 256, 0, stream>>>(word_emb, widx, bufX, 4096, 300, 364);
        gather_rows<<<blocks(32768LL * 64), 256, 0, stream>>>(char_emb, cidx, cemb, 32768, 64, 64);
        // char input projection: [32768,64] @ [256,64]^T + b
        launch_gemm(stream, cemb, 64, 0, cWih, 64, 0, 0, cb, nullptr, 0, 0,
                    cxw, 256, 0, 32768, 256, 64, 0, 1);
        hipMemsetAsync(hst, 0, 4096 * 64 * sizeof(float), stream);
        hipMemsetAsync(cst, 0, 4096 * 64 * sizeof(float), stream);
        for (int t = 0; t < 8; ++t) {
            launch_gemm(stream, hst, 64, 0, cWhh, 64, 0, 0, nullptr,
                        cxw + t * 256, 8 * 256, 0, gbuf, 256, 0, 4096, 256, 64, 0, 1);
            lstm_cell<<<blocks(4096 * 64), 256, 0, stream>>>(gbuf, hst, cst, 4096, 64, nullptr, 0);
        }
        copy_strided<<<blocks(4096 * 64), 256, 0, stream>>>(hst, 64, bufX, 364, 300, 4096, 64);
        // ctx input projections: [4096,364] @ [512,364]^T + b
        launch_gemm(stream, bufX, 364, 0, xWf, 364, 0, 0, xbf, nullptr, 0, 0,
                    xwF, 512, 0, 4096, 512, 364, 0, 1);
        launch_gemm(stream, bufX, 364, 0, xWb, 364, 0, 0, xbb, nullptr, 0, 0,
                    xwB, 512, 0, 4096, 512, 364, 0, 1);
        // forward scan
        hipMemsetAsync(hst, 0, 64 * 128 * sizeof(float), stream);
        hipMemsetAsync(cst, 0, 64 * 128 * sizeof(float), stream);
        for (int t = 0; t < 64; ++t) {
            launch_gemm(stream, hst, 128, 0, xUf, 128, 0, 0, nullptr,
                        xwF + t * 512, 64 * 512, 0, gbuf, 512, 0, 64, 512, 128, 0, 1);
            lstm_cell<<<blocks(64 * 128), 256, 0, stream>>>(gbuf, hst, cst, 64, 128,
                                                            o_fw + t * 128, 64 * 128);
        }
        // backward scan
        hipMemsetAsync(hst, 0, 64 * 128 * sizeof(float), stream);
        hipMemsetAsync(cst, 0, 64 * 128 * sizeof(float), stream);
        for (int t = 0; t < 64; ++t) {
            int l = 63 - t;
            launch_gemm(stream, hst, 128, 0, xUb, 128, 0, 0, nullptr,
                        xwB + l * 512, 64 * 512, 0, gbuf, 512, 0, 64, 512, 128, 0, 1);
            lstm_cell<<<blocks(64 * 128), 256, 0, stream>>>(gbuf, hst, cst, 64, 128,
                                                            o_bw + l * 128, 64 * 128);
        }
    };

    // ------------------- multi-perspective matching -------------------------
    auto mpm_run = [&](const float* h1f, const float* h1b, const float* h2f, const float* h2b,
                       const float* n1fv, const float* n1bv, const float* n2fv, const float* n2bv,
                       float* mout) {
        launch_gemm(stream, h1f, 128, 8192, h2f, 128, 8192, 0, nullptr, nullptr, 0, 0,
                    cmF, 64, 4096, 64, 64, 128, 0, 64);
        launch_gemm(stream, h1b, 128, 8192, h2b, 128, 8192, 0, nullptr, nullptr, 0, 0,
                    cmB, 64, 4096, 64, 64, 128, 0, 64);
        cm_normalize<<<blocks(64 * 64 * 64), 256, 0, stream>>>(cmF, n1fv, n2fv);
        cm_normalize<<<blocks(64 * 64 * 64), 256, 0, stream>>>(cmB, n1bv, n2bv);
        cm_rowsum<<<blocks(4096), 256, 0, stream>>>(cmF, rsF, 4096, 64);
        cm_rowsum<<<blocks(4096), 256, 0, stream>>>(cmB, rsB, 4096, 64);
        cm_argmax<<<blocks(4096), 256, 0, stream>>>(cmF, aidxF, 4096, 64);
        cm_argmax<<<blocks(4096), 256, 0, stream>>>(cmB, aidxB, 4096, 64);
        long long npl = 64 * 64 * 20;
        mpm_full<<<blocks(npl), 256, 0, stream>>>(h1f, h2f, mpmw + 0 * 2560, mout, 0);
        mpm_full<<<blocks(npl), 256, 0, stream>>>(h1b, h2b, mpmw + 1 * 2560, mout, 20);
        launch_gemm(stream, cmF, 64, 4096, h2f, 128, 8192, 1, nullptr, nullptr, 0, 0,
                    attb, 128, 8192, 64, 128, 64, 0, 64);
        mpm_att<<<blocks(npl), 256, 0, stream>>>(h1f, attb, rsF, mpmw + 2 * 2560, mout, 40);
        launch_gemm(stream, cmB, 64, 4096, h2b, 128, 8192, 1, nullptr, nullptr, 0, 0,
                    attb, 128, 8192, 64, 128, 64, 0, 64);
        mpm_att<<<blocks(npl), 256, 0, stream>>>(h1b, attb, rsB, mpmw + 3 * 2560, mout, 60);
        mpm_maxatt<<<blocks(npl), 256, 0, stream>>>(h1f, h2f, aidxF, mpmw + 4 * 2560, mout, 80);
        mpm_maxatt<<<blocks(npl), 256, 0, stream>>>(h1b, h2b, aidxB, mpmw + 5 * 2560, mout, 100);
        norms_hw<<<blocks(npl), 256, 0, stream>>>(h1f, mpmw + 6 * 2560, n1w);
        norms_hw<<<blocks(npl), 256, 0, stream>>>(h2f, mpmw + 6 * 2560, n2w);
        maxpool_wmma_k<<<dim3(4, 20, 64), 32, 0, stream>>>(h1f, h2f, mpmw + 6 * 2560,
                                                           n1w, n2w, mout, 120);
        norms_hw<<<blocks(npl), 256, 0, stream>>>(h1b, mpmw + 7 * 2560, n1w);
        norms_hw<<<blocks(npl), 256, 0, stream>>>(h2b, mpmw + 7 * 2560, n2w);
        maxpool_wmma_k<<<dim3(4, 20, 64), 32, 0, stream>>>(h1b, h2b, mpmw + 7 * 2560,
                                                           n1w, n2w, mout, 140);
    };

    // ------------------- aggregation BiLSTM (final states only) -------------
    auto agg_run = [&](const float* minput, int fcOffF, int fcOffB) {
        launch_gemm(stream, minput, 160, 0, aWf, 160, 0, 0, abf, nullptr, 0, 0,
                    xwF, 512, 0, 4096, 512, 160, 0, 1);
        launch_gemm(stream, minput, 160, 0, aWb, 160, 0, 0, abb, nullptr, 0, 0,
                    xwB, 512, 0, 4096, 512, 160, 0, 1);
        hipMemsetAsync(hst, 0, 64 * 128 * sizeof(float), stream);
        hipMemsetAsync(cst, 0, 64 * 128 * sizeof(float), stream);
        for (int t = 0; t < 64; ++t) {
            launch_gemm(stream, hst, 128, 0, aUf, 128, 0, 0, nullptr,
                        xwF + t * 512, 64 * 512, 0, gbuf, 512, 0, 64, 512, 128, 0, 1);
            lstm_cell<<<blocks(64 * 128), 256, 0, stream>>>(gbuf, hst, cst, 64, 128, nullptr, 0);
        }
        copy_strided<<<blocks(64 * 128), 256, 0, stream>>>(hst, 128, fcx, 512, fcOffF, 64, 128);
        hipMemsetAsync(hst, 0, 64 * 128 * sizeof(float), stream);
        hipMemsetAsync(cst, 0, 64 * 128 * sizeof(float), stream);
        for (int t = 0; t < 64; ++t) {
            int l = 63 - t;
            launch_gemm(stream, hst, 128, 0, aUb, 128, 0, 0, nullptr,
                        xwB + l * 512, 64 * 512, 0, gbuf, 512, 0, 64, 512, 128, 0, 1);
            lstm_cell<<<blocks(64 * 128), 256, 0, stream>>>(gbuf, hst, cst, 64, 128, nullptr, 0);
        }
        copy_strided<<<blocks(64 * 128), 256, 0, stream>>>(hst, 128, fcx, 512, fcOffB, 64, 128);
    };

    // =========================== pipeline ===================================
    encode(idxA, cidxA, p_fw, p_bw);
    encode(idxBc, cidxB, q_fw, q_bw);

    row_norm<<<blocks(4096), 256, 0, stream>>>(p_fw, nPF, 4096, 128);
    row_norm<<<blocks(4096), 256, 0, stream>>>(p_bw, nPB, 4096, 128);
    row_norm<<<blocks(4096), 256, 0, stream>>>(q_fw, nQF, 4096, 128);
    row_norm<<<blocks(4096), 256, 0, stream>>>(q_bw, nQB, 4096, 128);

    mpm_run(p_fw, p_bw, q_fw, q_bw, nPF, nPB, nQF, nQB, mp);
    mpm_run(q_fw, q_bw, p_fw, p_bw, nQF, nQB, nPF, nPB, mq);

    agg_run(mp, 0, 128);
    agg_run(mq, 256, 384);

    // fc1 with tanh, then fc2
    launch_gemm(stream, fcx, 512, 0, fc1W, 512, 0, 0, fc1b, nullptr, 0, 0,
                fc1o, 256, 0, 64, 256, 512, 1, 1);
    fc2_k<<<1, 128, 0, stream>>>(fc1o, fc2W, fc2b, out);
}
